// Model_58600533787271
// MI455X (gfx1250) — matrix operand order repaired and checked by reading, not compiled, not run on hardware
//
#include <hip/hip_runtime.h>
#include <hip/hip_bf16.h>
#include <cstdint>

// ---------------------------------------------------------------------------
// Shapes
// ---------------------------------------------------------------------------
#define D_MODEL 256
#define VOCAB   50257
#define VPAD    50272          // VOCAB rounded up to multiple of 32
#define NTOK    4096           // B(4) * S(1024)
#define SEQ     1024
#define IN_DIM  64
#define OUT_DIM 64
#define NCHUNK  (VPAD / 32)    // 1571 vocab chunks of 32

typedef __attribute__((ext_vector_type(16))) __bf16 bf16x16;
typedef __attribute__((ext_vector_type(8)))  __bf16 bf16x8;
typedef __attribute__((ext_vector_type(8)))  float  f32x8;
typedef __attribute__((ext_vector_type(4)))  unsigned int u32x4;
typedef __attribute__((ext_vector_type(4)))  int v4i;

__device__ __forceinline__ __bf16 f2bf(float f) { return (__bf16)f; }

// CDNA5 async global->LDS copies (ASYNCcnt-tracked), guarded so the file
// compiles even if the toolchain lacks the builtins.
#if defined(__gfx1250__) && \
    __has_builtin(__builtin_amdgcn_global_load_async_to_lds_b128) && \
    __has_builtin(__builtin_amdgcn_s_wait_asynccnt)
  #define USE_ASYNC 1
#else
  #define USE_ASYNC 0
#endif

// ---------------------------------------------------------------------------
// Kernel 0a: embedding f32 -> bf16, row-major, zero-padded to VPAD rows
// ---------------------------------------------------------------------------
__global__ void k_emb_row(const float* __restrict__ emb, __bf16* __restrict__ embr)
{
    size_t idx = (size_t)blockIdx.x * blockDim.x + threadIdx.x;
    if (idx >= (size_t)VPAD * D_MODEL) return;
    size_t v = idx / D_MODEL;
    size_t d = idx % D_MODEL;
    float val = (v < VOCAB) ? emb[v * D_MODEL + d] : 0.0f;
    embr[idx] = f2bf(val);
}

// ---------------------------------------------------------------------------
// Kernel 0b: embedding f32 -> bf16 TRANSPOSED  embT[d][v], v padded to VPAD
// ---------------------------------------------------------------------------
__global__ void k_emb_t(const float* __restrict__ emb, __bf16* __restrict__ embT)
{
    size_t idx = (size_t)blockIdx.x * blockDim.x + threadIdx.x;
    if (idx >= (size_t)D_MODEL * VPAD) return;
    size_t d  = idx / VPAD;
    size_t vc = idx % VPAD;
    float val = (vc < VOCAB) ? emb[vc * D_MODEL + d] : 0.0f;
    embT[idx] = f2bf(val);
}

// ---------------------------------------------------------------------------
// Kernel 1: encoder GEMM + positional emb + LayerNorm, writes h in bf16.
// ---------------------------------------------------------------------------
__global__ __launch_bounds__(256)
void k_encoder(const float* __restrict__ x, const float* __restrict__ wpe,
               const float* __restrict__ enc_w, const float* __restrict__ enc_b,
               const float* __restrict__ ln_g, const float* __restrict__ ln_b,
               __bf16* __restrict__ hbf)
{
    __shared__ float xs[IN_DIM];
    __shared__ float red[8];
    __shared__ float stat;

    const int row  = blockIdx.x;
    const int d    = threadIdx.x;
    const int lane = d & 31;
    const int wid  = d >> 5;

    if (d < IN_DIM) xs[d] = x[(size_t)row * IN_DIM + d];
    __syncthreads();

    float acc = enc_b[d];
    #pragma unroll
    for (int i = 0; i < IN_DIM; ++i)
        acc = fmaf(xs[i], enc_w[(size_t)d * IN_DIM + i], acc);
    acc += wpe[(size_t)(row & (SEQ - 1)) * D_MODEL + d];

    // mean
    float s = acc;
    #pragma unroll
    for (int off = 1; off < 32; off <<= 1) s += __shfl_xor(s, off, 32);
    if (lane == 0) red[wid] = s;
    __syncthreads();
    if (d == 0) {
        float t = 0.f;
        #pragma unroll
        for (int w = 0; w < 8; ++w) t += red[w];
        stat = t * (1.0f / D_MODEL);
    }
    __syncthreads();
    const float mu = stat;
    const float dv = acc - mu;
    __syncthreads();

    // variance
    float s2 = dv * dv;
    #pragma unroll
    for (int off = 1; off < 32; off <<= 1) s2 += __shfl_xor(s2, off, 32);
    if (lane == 0) red[wid] = s2;
    __syncthreads();
    if (d == 0) {
        float t = 0.f;
        #pragma unroll
        for (int w = 0; w < 8; ++w) t += red[w];
        stat = t * (1.0f / D_MODEL);
    }
    __syncthreads();

    const float r = rsqrtf(stat + 1e-5f);
    hbf[(size_t)row * D_MODEL + d] = f2bf(dv * r * ln_g[d] + ln_b[d]);
}

// ---------------------------------------------------------------------------
// Per-chunk staging into LDS (both layouts), 128 threads x 8 x 16B each.
// ---------------------------------------------------------------------------
__device__ __forceinline__ void stage_chunk(__bf16* eb, __bf16* tb,
                                            const __bf16* __restrict__ embr,
                                            const __bf16* __restrict__ embT,
                                            int vb, int tid)
{
    const __bf16* gs = embr + (size_t)vb * D_MODEL;   // contiguous 16 KB
    #pragma unroll
    for (int r = 0; r < 8; ++r) {
        const int p   = tid + 128 * r;   // 0..1023 pieces of 16B
        const int row = p >> 2;          // 0..255
        const int c4  = p & 3;           // 16B sub-piece of a 64B row
#if USE_ASYNC
        __builtin_amdgcn_global_load_async_to_lds_b128(
            (v4i*)(gs + p * 8),
            (v4i*)(eb + p * 8), 0, 0);
        __builtin_amdgcn_global_load_async_to_lds_b128(
            (v4i*)(embT + (size_t)row * VPAD + vb + c4 * 8),
            (v4i*)(tb + row * 32 + c4 * 8), 0, 0);
#else
        *(u32x4*)(eb + p * 8) = *(const u32x4*)(gs + p * 8);
        *(u32x4*)(tb + row * 32 + c4 * 8) =
            *(const u32x4*)(embT + (size_t)row * VPAD + vb + c4 * 8);
#endif
    }
}

// ---------------------------------------------------------------------------
// Kernel 2: fused  logits -> online softmax -> quantize  (flash style).
// ---------------------------------------------------------------------------
__global__ __launch_bounds__(128, 1)
void k_fused(const __bf16* __restrict__ hbf,
             const __bf16* __restrict__ embr,   // [VPAD][256] bf16
             const __bf16* __restrict__ embT,   // [256][VPAD] bf16
             float* __restrict__ qout,          // [NTOK][256] f32
             int* __restrict__ amax)            // [NTOK]
{
    extern __shared__ char smem[];
    __bf16* ebuf = (__bf16*)smem;                 // [2][32][256]  32 KB
    __bf16* tbuf = (__bf16*)(smem + 32768);       // [2][256][32]  32 KB
    __bf16* pbuf = (__bf16*)(smem + 65536);       // [4][16][32]    4 KB

    const int tid  = threadIdx.x;
    const int lane = tid & 31;
    const int wave = tid >> 5;
    const int row0 = blockIdx.x * 64 + wave * 16;
    const int m    = lane & 15;     // row (A) or column (B/C) index
    const int hh   = lane >> 4;     // half-wave selector

    // ---- resident A fragments of the 16x256 h tile -------------------------
    bf16x16 a[8];
    {
        const __bf16* hrow = hbf + (size_t)(row0 + m) * D_MODEL;
        #pragma unroll
        for (int s = 0; s < 8; ++s) {
            bf16x8 lo = *(const bf16x8*)(hrow + 32 * s + hh * 16);
            bf16x8 hi = *(const bf16x8*)(hrow + 32 * s + hh * 16 + 8);
            bf16x16 t;
            #pragma unroll
            for (int j = 0; j < 8; ++j) { t[j] = lo[j]; t[8 + j] = hi[j]; }
            a[s] = t;
        }
    }

    // ---- accumulators ------------------------------------------------------
    f32x8 q[16];
    #pragma unroll
    for (int t = 0; t < 16; ++t)
        #pragma unroll
        for (int v = 0; v < 8; ++v) q[t][v] = 0.0f;

    f32x8 mrun, lrun, bestv;
    int bi[8];
    #pragma unroll
    for (int v = 0; v < 8; ++v) {
        mrun[v] = -3.0e38f; lrun[v] = 0.0f; bestv[v] = -3.0e38f; bi[v] = 0;
    }

    __bf16* pw = pbuf + wave * (16 * 32);

    // prologue: stage chunks 0 and 1
    stage_chunk(ebuf,            tbuf,            embr, embT, 0,  tid);
    stage_chunk(ebuf + 32 * 256, tbuf + 256 * 32, embr, embT, 32, tid);

    #pragma unroll 1
    for (int c = 0; c < NCHUNK; ++c) {
        const int vb = c * 32;
        __bf16* eb = ebuf + (c & 1) * (32 * 256);
        __bf16* tb = tbuf + (c & 1) * (256 * 32);

#if USE_ASYNC
        // chunk c's 16 async copies done; chunk c+1's 16 may stay in flight
        if (c + 1 < NCHUNK) __builtin_amdgcn_s_wait_asynccnt(16);
        else                __builtin_amdgcn_s_wait_asynccnt(0);
#endif
        __syncthreads();   // staged chunk visible to all waves

        // ---- S = (h/16) * E^T, software-pipelined B fragments ---------------
        f32x8 c0, c1;
        #pragma unroll
        for (int v = 0; v < 8; ++v) { c0[v] = 0.0f; c1[v] = 0.0f; }

        const __bf16* ebm0 = eb + m * D_MODEL + hh * 16;
        const __bf16* ebm1 = eb + (m + 16) * D_MODEL + hh * 16;
        bf16x16 B0 = *(const bf16x16*)(ebm0);
        bf16x16 B1 = *(const bf16x16*)(ebm1);
        #pragma unroll
        for (int s = 0; s < 8; ++s) {
            bf16x16 N0 = B0, N1 = B1;
            if (s < 7) {
                N0 = *(const bf16x16*)(ebm0 + 32 * (s + 1));
                N1 = *(const bf16x16*)(ebm1 + 32 * (s + 1));
            }
            c0 = __builtin_amdgcn_wmma_f32_16x16x32_bf16(false, a[s], false, B0,
                                                         (short)0, c0, false, false);
            c1 = __builtin_amdgcn_wmma_f32_16x16x32_bf16(false, a[s], false, B1,
                                                         (short)0, c1, false, false);
            B0 = N0; B1 = N1;
        }

        // scale logits by 1/sqrt(256)
        #pragma unroll
        for (int v = 0; v < 8; ++v) { c0[v] *= 0.0625f; c1[v] *= 0.0625f; }

        // mask padded vocab columns in the tail chunk
        if (vb + 32 > VOCAB) {
            const bool k0 = (vb + m)      >= VOCAB;
            const bool k1 = (vb + 16 + m) >= VOCAB;
            #pragma unroll
            for (int v = 0; v < 8; ++v) {
                if (k0) c0[v] = -3.0e38f;
                if (k1) c1[v] = -3.0e38f;
            }
        }

        // running argmax
        #pragma unroll
        for (int v = 0; v < 8; ++v) {
            if (c0[v] > bestv[v]) { bestv[v] = c0[v]; bi[v] = vb + m; }
            if (c1[v] > bestv[v]) { bestv[v] = c1[v]; bi[v] = vb + 16 + m; }
        }

        // ---- online softmax ------------------------------------------------
        f32x8 cmax;
        #pragma unroll
        for (int v = 0; v < 8; ++v) cmax[v] = fmaxf(c0[v], c1[v]);
        #pragma unroll
        for (int off = 1; off < 16; off <<= 1)
            #pragma unroll
            for (int v = 0; v < 8; ++v)
                cmax[v] = fmaxf(cmax[v], __shfl_xor(cmax[v], off, 32));

        bool newmax = false;
        #pragma unroll
        for (int v = 0; v < 8; ++v) newmax |= (cmax[v] > mrun[v]);

        f32x8 mnew, p0, p1, rs;
        #pragma unroll
        for (int v = 0; v < 8; ++v) {
            mnew[v] = fmaxf(mrun[v], cmax[v]);
            p0[v]   = __expf(c0[v] - mnew[v]);
            p1[v]   = __expf(c1[v] - mnew[v]);
            rs[v]   = p0[v] + p1[v];
        }
        #pragma unroll
        for (int off = 1; off < 16; off <<= 1)
            #pragma unroll
            for (int v = 0; v < 8; ++v) rs[v] += __shfl_xor(rs[v], off, 32);

        if (__any(newmax)) {
            // a row's max moved: rescale the running sum and Q accumulators
            f32x8 corr;
            #pragma unroll
            for (int v = 0; v < 8; ++v) corr[v] = __expf(mrun[v] - mnew[v]);
            #pragma unroll
            for (int v = 0; v < 8; ++v) lrun[v] = lrun[v] * corr[v] + rs[v];
            #pragma unroll
            for (int t = 0; t < 16; ++t) q[t] *= corr;
        } else {
            #pragma unroll
            for (int v = 0; v < 8; ++v) lrun[v] += rs[v];
        }
        #pragma unroll
        for (int v = 0; v < 8; ++v) mrun[v] = mnew[v];

        // ---- transpose P (C layout -> A fragment) through LDS --------------
        #pragma unroll
        for (int v = 0; v < 8; ++v) {
            pw[(v + 8 * hh) * 32 + m]      = f2bf(p0[v]);
            pw[(v + 8 * hh) * 32 + m + 16] = f2bf(p1[v]);
        }
        asm volatile("s_wait_dscnt 0x0" ::: "memory");  // wave-internal DS RAW

        bf16x16 pa;
        {
            bf16x8 plo = *(const bf16x8*)(pw + m * 32 + hh * 16);
            bf16x8 phi = *(const bf16x8*)(pw + m * 32 + hh * 16 + 8);
            #pragma unroll
            for (int j = 0; j < 8; ++j) { pa[j] = plo[j]; pa[8 + j] = phi[j]; }
        }

        // ---- Q += P * E_chunk, software-pipelined B fragments ---------------
        const __bf16* tbm = tb + m * 32 + hh * 16;
        bf16x16 Bq = *(const bf16x16*)(tbm);
        #pragma unroll
        for (int t = 0; t < 16; ++t) {
            bf16x16 Nq = Bq;
            if (t < 15) Nq = *(const bf16x16*)(tbm + (t + 1) * (16 * 32));
            q[t] = __builtin_amdgcn_wmma_f32_16x16x32_bf16(false, pa, false, Bq,
                                                           (short)0, q[t], false, false);
            Bq = Nq;
        }

        __syncthreads();   // all waves done reading buf (c&1)
        if (c + 2 < NCHUNK)
            stage_chunk(ebuf + (c & 1) * (32 * 256), tbuf + (c & 1) * (256 * 32),
                        embr, embT, vb + 64, tid);
    }

    // ---- finalize: divide by softmax denominator, write quantize -----------
    f32x8 inv;
    #pragma unroll
    for (int v = 0; v < 8; ++v) inv[v] = 1.0f / lrun[v];

    #pragma unroll
    for (int t = 0; t < 16; ++t)
        #pragma unroll
        for (int v = 0; v < 8; ++v)
            qout[(size_t)(row0 + v + 8 * hh) * D_MODEL + 16 * t + m] = q[t][v] * inv[v];

    // ---- argmax cross-lane reduction ---------------------------------------
    #pragma unroll
    for (int off = 1; off < 16; off <<= 1) {
        #pragma unroll
        for (int v = 0; v < 8; ++v) {
            float ov = __shfl_xor(bestv[v], off, 32);
            int   oi = __shfl_xor(bi[v], off, 32);
            if (ov > bestv[v] || (ov == bestv[v] && oi < bi[v])) {
                bestv[v] = ov; bi[v] = oi;
            }
        }
    }
    if (m == 0) {
        #pragma unroll
        for (int v = 0; v < 8; ++v) amax[row0 + v + 8 * hh] = bi[v];
    }
}

// ---------------------------------------------------------------------------
// Kernel 3: decoder  out[row][o] = quantize[row] . dec_w[o] + dec_b[o]
// ---------------------------------------------------------------------------
__global__ __launch_bounds__(64)
void k_decoder(const float* __restrict__ qbuf, const float* __restrict__ dec_w,
               const float* __restrict__ dec_b, float* __restrict__ out)
{
    __shared__ float qs[D_MODEL];
    const int row = blockIdx.x;
    const int o   = threadIdx.x;
    for (int i = o; i < D_MODEL; i += 64) qs[i] = qbuf[(size_t)row * D_MODEL + i];
    __syncthreads();
    float acc = dec_b[o];
    #pragma unroll 8
    for (int d = 0; d < D_MODEL; ++d)
        acc = fmaf(qs[d], dec_w[(size_t)o * D_MODEL + d], acc);
    out[(size_t)row * OUT_DIM + o] = acc;
}

// ---------------------------------------------------------------------------
// Launch
// ---------------------------------------------------------------------------
extern "C" void kernel_launch(void* const* d_in, const int* in_sizes, int n_in,
                              void* d_out, int out_size, void* d_ws, size_t ws_size,
                              hipStream_t stream)
{
    const float* x     = (const float*)d_in[0];
    const float* emb   = (const float*)d_in[1];
    const float* wpe   = (const float*)d_in[2];
    const float* enc_w = (const float*)d_in[3];
    const float* enc_b = (const float*)d_in[4];
    const float* ln_g  = (const float*)d_in[5];
    const float* ln_b  = (const float*)d_in[6];
    const float* dec_w = (const float*)d_in[7];
    const float* dec_b = (const float*)d_in[8];

    // workspace layout (~57.8 MB total)
    __bf16* embr = (__bf16*)d_ws;                             // VPAD*256 bf16
    __bf16* embT = embr + (size_t)VPAD * D_MODEL;             // 256*VPAD bf16
    __bf16* hbf  = embT + (size_t)D_MODEL * VPAD;             // NTOK*256 bf16
    float*  qbuf = (float*)(hbf + (size_t)NTOK * D_MODEL);    // NTOK*256 f32

    float* out  = (float*)d_out;                              // NTOK*OUT_DIM f32
    int*   amax = (int*)d_out + (size_t)NTOK * OUT_DIM;       // NTOK i32

    const size_t nconv = (size_t)VPAD * D_MODEL;
    k_emb_row<<<(unsigned)((nconv + 255) / 256), 256, 0, stream>>>(emb, embr);
    k_emb_t  <<<(unsigned)((nconv + 255) / 256), 256, 0, stream>>>(emb, embT);
    k_encoder<<<NTOK, 256, 0, stream>>>(x, wpe, enc_w, enc_b, ln_g, ln_b, hbf);
    k_fused  <<<NTOK / 64, 128, 67584, stream>>>(hbf, embr, embT, qbuf, amax);
    k_decoder<<<NTOK, 64, 0, stream>>>(qbuf, dec_w, dec_b, out);
}
